// SegSmall_Features_Discriminotor_3642132267204
// MI455X (gfx1250) — matrix operand order repaired and checked by reading, not compiled, not run on hardware
//
#include <hip/hip_runtime.h>
#include <hip/hip_bf16.h>

typedef __attribute__((ext_vector_type(16))) _Float16 v16h;
typedef __attribute__((ext_vector_type(8)))  _Float16 h8;
typedef __attribute__((ext_vector_type(4)))  _Float16 h4;
typedef __attribute__((ext_vector_type(8)))  float    v8f;
typedef __attribute__((ext_vector_type(4)))  int      v4i;

#define EPSF 1e-5f

// ---- optional CDNA5 async global->LDS path (graceful fallback) ----
#if defined(__has_builtin)
#if __has_builtin(__builtin_amdgcn_global_load_async_to_lds_b128)
#define PT_ASYNC_LDS 1
#endif
#if __has_builtin(__builtin_amdgcn_s_wait_asynccnt)
#define PT_WAIT_ASYNC_BUILTIN 1
#endif
#endif

#ifdef PT_ASYNC_LDS
typedef __attribute__((address_space(1))) v4i* gv4i_p;
typedef __attribute__((address_space(3))) v4i* lv4i_p;
#endif

__device__ __forceinline__ void async_copy_f32(const float* __restrict__ g,
                                               float* __restrict__ l,
                                               int nfloats, int t) {
#ifdef PT_ASYNC_LDS
  for (int i = t * 4; i < nfloats; i += 128 * 4) {
    __builtin_amdgcn_global_load_async_to_lds_b128(
        (gv4i_p)(v4i*)(const_cast<float*>(g) + i),
        (lv4i_p)(v4i*)(l + i), 0, 0);
  }
#else
  for (int i = t; i < nfloats; i += 128) l[i] = g[i];
#endif
}

__device__ __forceinline__ void wait_async_lds() {
#ifdef PT_ASYNC_LDS
#ifdef PT_WAIT_ASYNC_BUILTIN
  __builtin_amdgcn_s_wait_asynccnt(0);
#else
  asm volatile("s_wait_asynccnt 0" ::: "memory");
#endif
#endif
}

// ---------------------------------------------------------------------------
// Pack W [KG, COUT] f32 into per-lane WMMA B-fragment layout (f16):
// Wpk[((s*NT + nt)*32 + lane)*16 + e] = W[(s*32 + kkb(lane) + e + (e>=8 ? 8 : 0))*COUT + nt*16 + (lane&15)]
// so the conv kernel fetches each B fragment as one contiguous 32B load.
// ---------------------------------------------------------------------------
__global__ void pack_w_kernel(const float* __restrict__ W,
                              _Float16* __restrict__ Wpk, int KG, int COUT) {
  int NT = COUT >> 4;
  int total = (KG >> 5) * NT * 32;
  int idx = blockIdx.x * blockDim.x + threadIdx.x;
  if (idx >= total) return;
  int l = idx & 31;
  int tid = idx >> 5;
  int nt = tid % NT;
  int s = tid / NT;
  int kkb = (l < 16) ? 0 : 8;
  int col = nt * 16 + (l & 15);
  _Float16* dst = Wpk + (size_t)idx * 16;
#pragma unroll
  for (int e = 0; e < 16; ++e)
    dst[e] = (_Float16)W[(size_t)(s * 32 + kkb + e + (e >= 8 ? 8 : 0)) * COUT + col];
}

// ---------------------------------------------------------------------------
// Brute-force KNN: one thread per output point, register insertion sort.
// ---------------------------------------------------------------------------
template <int K>
__global__ void knn_kernel(const float* __restrict__ pts,   // [B,N,3]
                           const float* __restrict__ nxt,   // [B,M,3]
                           int* __restrict__ knn,           // [B,M,K]
                           int B, int N, int M) {
  int gid = blockIdx.x * blockDim.x + threadIdx.x;
  if (gid >= B * M) return;
  int b = gid / M;
  float qx = nxt[gid * 3 + 0], qy = nxt[gid * 3 + 1], qz = nxt[gid * 3 + 2];
  float bd[K];
  int bi[K];
#pragma unroll
  for (int s = 0; s < K; ++s) { bd[s] = 3.4e38f; bi[s] = 0; }
  const float* P = pts + (size_t)b * N * 3;
  for (int j = 0; j < N; ++j) {
    float dx = P[j * 3 + 0] - qx;
    float dy = P[j * 3 + 1] - qy;
    float dz = P[j * 3 + 2] - qz;
    float d2 = dx * dx + dy * dy + dz * dz;
    if (d2 < bd[K - 1]) {
      bd[K - 1] = d2; bi[K - 1] = j;
#pragma unroll
      for (int s = K - 1; s > 0; --s) {
        if (bd[s] < bd[s - 1]) {
          float td = bd[s]; bd[s] = bd[s - 1]; bd[s - 1] = td;
          int   ti = bi[s]; bi[s] = bi[s - 1]; bi[s - 1] = ti;
        }
      }
    }
  }
#pragma unroll
  for (int s = 0; s < K; ++s) knn[(size_t)gid * K + s] = bi[s];
}

// ---------------------------------------------------------------------------
// Fused point-conv tile kernel (16 output points / workgroup, 4 wave32).
// ---------------------------------------------------------------------------
template <int CIN, int COUT, int K>
__global__ __launch_bounds__(128) void ptconv_tile_kernel(
    const float* __restrict__ feat,      // [B,N,CIN]
    const float* __restrict__ pts,       // [B,N,3]
    const float* __restrict__ nxt,       // [B,M,3]
    const int* __restrict__ knn,         // [B,M,K]
    const _Float16* __restrict__ Wh,     // packed B fragments (see pack_w_kernel)
    const float* __restrict__ cent,      // [3,16]
    const float* __restrict__ l1w, const float* __restrict__ l1b,  // [32,48],[32]
    const float* __restrict__ l2w, const float* __restrict__ l2b,  // [16,32],[16]
    const float* __restrict__ l3w, const float* __restrict__ l3b,  // [16,16],[16]
    float* __restrict__ out,             // [B,M,COUT]
    int B, int N, int M) {
  constexpr int TILE = 16;
  constexpr int NT = COUT / 16;          // 16-wide N tiles (3 or 6)
  constexpr int STEPS = (CIN * 16) / 32; // GEMM-K chunks of 32

  __shared__ float s_cent[48];
  __shared__ float s_l1w[32 * 48];
  __shared__ float s_l1b[32];
  __shared__ float s_l2w[16 * 32];
  __shared__ float s_l2b[16];
  __shared__ float s_l3w[16 * 16];
  __shared__ float s_l3b[16];
  __shared__ float s_h[TILE][8][16];     // per-(point,k) MLP output
  __shared__ float s_r2[TILE][8];
  __shared__ _Float16 s_gf[TILE][K][CIN];
  __shared__ _Float16 s_A[TILE][32];     // staged A chunk (16 x 32 f16)

  const int t = threadIdx.x;
  const int lane = t & 31;
  const int wv = t >> 5;
  const int tile0 = blockIdx.x * TILE;   // base row in [0, B*M)

  // cooperative preload of MLP params into LDS (async DMA if available)
  async_copy_f32(cent, s_cent, 48, t);
  async_copy_f32(l1w, s_l1w, 32 * 48, t);
  async_copy_f32(l1b, s_l1b, 32, t);
  async_copy_f32(l2w, s_l2w, 16 * 32, t);
  async_copy_f32(l2b, s_l2b, 16, t);
  async_copy_f32(l3w, s_l3w, 16 * 16, t);
  async_copy_f32(l3b, s_l3b, 16, t);
  wait_async_lds();

  // ------------------ phase 1a: rel + feature gather ------------------
  const bool act = (t < TILE * K);
  const int ml = t / K;
  const int kk0 = t % K;
  float rx = 0.f, ry = 0.f, rz = 0.f;
  if (act) {
    int gm = tile0 + ml;
    int b = gm / M;
    int nb = knn[(size_t)gm * K + kk0];
    const float* q = nxt + (size_t)gm * 3;
    const float* p = pts + ((size_t)b * N + nb) * 3;
    rx = p[0] - q[0]; ry = p[1] - q[1]; rz = p[2] - q[2];
    s_r2[ml][kk0] = rx * rx + ry * ry + rz * rz;
    const float* fr = feat + ((size_t)b * N + nb) * CIN;
#pragma unroll 4
    for (int c = 0; c < CIN; c += 4) {
      float4 f4 = *(const float4*)(fr + c);
      h4 hv = {(_Float16)f4.x, (_Float16)f4.y, (_Float16)f4.z, (_Float16)f4.w};
      *(h4*)&s_gf[ml][kk0][c] = hv;
    }
  }
  __syncthreads();

  // ------------------ phase 1b: normalize + MLP -> h ------------------
  if (act) {
    float mx = 0.f;
#pragma unroll
    for (int k2 = 0; k2 < K; ++k2) mx = fmaxf(mx, s_r2[ml][k2]);
    float maxi = sqrtf(mx);
    if (maxi == 0.f) maxi = 1.f;
    float inv = 1.f / maxi;
    float r3[3] = {rx * inv, ry * inv, rz * inv};

    float d[48];
#pragma unroll
    for (int j = 0; j < 48; j += 4) {
      float4 cv = *(const float4*)&s_cent[j];
      float rj = r3[j >> 4];
      d[j + 0] = rj - cv.x; d[j + 1] = rj - cv.y;
      d[j + 2] = rj - cv.z; d[j + 3] = rj - cv.w;
    }
    float h1[32];
#pragma unroll
    for (int o = 0; o < 32; ++o) {
      float a = s_l1b[o];
#pragma unroll
      for (int j = 0; j < 48; j += 4) {
        float4 wv = *(const float4*)&s_l1w[o * 48 + j];
        a += d[j] * wv.x + d[j + 1] * wv.y + d[j + 2] * wv.z + d[j + 3] * wv.w;
      }
      h1[o] = fmaxf(a, 0.f);
    }
    float h2[16];
#pragma unroll
    for (int o = 0; o < 16; ++o) {
      float a = s_l2b[o];
#pragma unroll
      for (int j = 0; j < 32; j += 4) {
        float4 wv = *(const float4*)&s_l2w[o * 32 + j];
        a += h1[j] * wv.x + h1[j + 1] * wv.y + h1[j + 2] * wv.z + h1[j + 3] * wv.w;
      }
      h2[o] = fmaxf(a, 0.f);
    }
#pragma unroll
    for (int o = 0; o < 16; ++o) {
      float a = s_l3b[o];
#pragma unroll
      for (int j = 0; j < 16; j += 4) {
        float4 wv = *(const float4*)&s_l3w[o * 16 + j];
        a += h2[j] * wv.x + h2[j + 1] * wv.y + h2[j + 2] * wv.z + h2[j + 3] * wv.w;
      }
      s_h[ml][kk0][o] = fmaxf(a, 0.f);
    }
  }
  __syncthreads();

  // ------------------ phase 2: WMMA over K-chunks ------------------
  v8f acc0 = {};
  v8f acc1 = {};
  const int nt0 = wv;      // N-tiles owned by this wave
  const int nt1 = wv + 4;
  // A fragment lane map (ISA 16-bit A 16x32 layout)
  const int arow = lane & 15;
  const int akb = (lane < 16) ? 0 : 8;
  const int ocol = lane & 15;  // output column within tile (C/D layout)

  // per-thread A-build coordinates: 4 consecutive kk in one row, one channel
  const int i0 = t * 4;
  const int am = i0 >> 5;
  const int kkb4 = i0 & 31;
  const int cofs = kkb4 >> 4;  // 0 or 1 within the 2-channel chunk
  const int n0 = kkb4 & 15;

  for (int s = 0; s < STEPS; ++s) {
    // build A chunk f[m, kk] = sum_k gf[m,k,c] * h[m,k,n]
    {
      const int c = (s << 1) + cofs;
      float a0 = 0.f, a1 = 0.f, a2 = 0.f, a3 = 0.f;
#pragma unroll
      for (int k2 = 0; k2 < K; ++k2) {
        float g = (float)s_gf[am][k2][c];
        float4 hv = *(const float4*)&s_h[am][k2][n0];
        a0 += g * hv.x; a1 += g * hv.y; a2 += g * hv.z; a3 += g * hv.w;
      }
      h4 av = {(_Float16)a0, (_Float16)a1, (_Float16)a2, (_Float16)a3};
      *(h4*)&s_A[am][kkb4] = av;
    }
    __syncthreads();

    // A fragment: two contiguous 16B LDS loads
    h8 alo = *(const h8*)&s_A[arow][akb];
    h8 ahi = *(const h8*)&s_A[arow][akb + 16];
    v16h af = __builtin_shufflevector(alo, ahi, 0, 1, 2, 3, 4, 5, 6, 7, 8, 9,
                                      10, 11, 12, 13, 14, 15);

    // packed B fragments: one 32B load per tile
    const _Float16* wbase = Wh + ((size_t)(s * NT) * 32 + lane) * 16;
    if (s + 1 < STEPS)
      __builtin_prefetch(Wh + ((size_t)((s + 1) * NT) * 32 + lane) * 16, 0, 0);

    if (nt0 < NT) {
      v16h bf = *(const v16h*)(wbase + (size_t)nt0 * 512);
      acc0 = __builtin_amdgcn_wmma_f32_16x16x32_f16(
          false, af, false, bf, (short)0, acc0, false, false);
    }
    if (nt1 < NT) {
      v16h bf = *(const v16h*)(wbase + (size_t)nt1 * 512);
      acc1 = __builtin_amdgcn_wmma_f32_16x16x32_f16(
          false, af, false, bf, (short)0, acc1, false, false);
    }
    __syncthreads();
  }

  // ------------------ epilogue: scale 1/K, store C tiles ------------------
  const float scale = 1.0f / (float)K;
  if (nt0 < NT) {
#pragma unroll
    for (int r = 0; r < 8; ++r) {
      int gm = tile0 + r + ((lane < 16) ? 0 : 8);
      out[(size_t)gm * COUT + nt0 * 16 + ocol] = acc0[r] * scale;
    }
  }
  if (nt1 < NT) {
#pragma unroll
    for (int r = 0; r < 8; ++r) {
      int gm = tile0 + r + ((lane < 16) ? 0 : 8);
      out[(size_t)gm * COUT + nt1 * 16 + ocol] = acc1[r] * scale;
    }
  }
}

// ---------------------------------------------------------------------------
// BatchNorm (training-mode) statistics: one block per channel.
// ---------------------------------------------------------------------------
__global__ void bn_stats_kernel(const float* __restrict__ x,
                                float* __restrict__ stats, int rows, int C) {
  int c = blockIdx.x;
  float s = 0.f, s2 = 0.f;
  for (int r = threadIdx.x; r < rows; r += 256) {
    float v = x[(size_t)r * C + c];
    s += v;
    s2 += v * v;
  }
  __shared__ float sh[256], sh2[256];
  sh[threadIdx.x] = s;
  sh2[threadIdx.x] = s2;
  __syncthreads();
  for (int o = 128; o > 0; o >>= 1) {
    if (threadIdx.x < o) {
      sh[threadIdx.x] += sh[threadIdx.x + o];
      sh2[threadIdx.x] += sh2[threadIdx.x + o];
    }
    __syncthreads();
  }
  if (threadIdx.x == 0) {
    float mu = sh[0] / rows;
    stats[2 * c] = mu;
    stats[2 * c + 1] = sh2[0] / rows - mu * mu;
  }
}

// ---------------------------------------------------------------------------
// y[:, :Cc] = relu(bn(conv)), y[:, Cc:Cc+Cs] = skip   (Cs may be 0)
// ---------------------------------------------------------------------------
__global__ void bn_apply_concat_kernel(const float* __restrict__ conv,
                                       const float* __restrict__ stats,
                                       const float* __restrict__ g,
                                       const float* __restrict__ bb,
                                       const float* __restrict__ skip,
                                       float* __restrict__ dst, int rows,
                                       int Cc, int Cs) {
  int Ct = Cc + Cs;
  int i = blockIdx.x * blockDim.x + threadIdx.x;
  if (i >= rows * Ct) return;
  int r = i / Ct, c = i % Ct;
  float o;
  if (c < Cc) {
    float mu = stats[2 * c], var = stats[2 * c + 1];
    float v = conv[(size_t)r * Cc + c];
    o = fmaxf(g[c] * (v - mu) * rsqrtf(var + EPSF) + bb[c], 0.f);
  } else {
    o = skip[(size_t)r * Cs + (c - Cc)];
  }
  dst[(size_t)r * Ct + c] = o;
}

// ---------------------------------------------------------------------------
// Output heads.  d_out: [rows*13] xout flat, then [rows] cout flat.
// ---------------------------------------------------------------------------
__global__ void head_kernel(const float* __restrict__ x1d,
                            const float* __restrict__ fw,
                            const float* __restrict__ fb,
                            const float* __restrict__ cw,
                            const float* __restrict__ cb,
                            float* __restrict__ out, int rows) {
  int r = blockIdx.x * blockDim.x + threadIdx.x;
  if (r >= rows) return;
  float f[48];
#pragma unroll
  for (int c = 0; c < 48; c += 4) {
    float4 v = *(const float4*)(x1d + (size_t)r * 48 + c);
    f[c] = v.x; f[c + 1] = v.y; f[c + 2] = v.z; f[c + 3] = v.w;
  }
#pragma unroll
  for (int j = 0; j < 13; ++j) {
    float a = fb[j];
#pragma unroll
    for (int c = 0; c < 48; c += 4) {
      float4 wv = *(const float4*)(fw + j * 48 + c);
      a += f[c] * wv.x + f[c + 1] * wv.y + f[c + 2] * wv.z + f[c + 3] * wv.w;
    }
    out[(size_t)r * 13 + j] = a;
  }
  float a = cb[0];
#pragma unroll
  for (int c = 0; c < 48; c += 4) {
    float4 wv = *(const float4*)(cw + c);
    a += f[c] * wv.x + f[c + 1] * wv.y + f[c + 2] * wv.z + f[c + 3] * wv.w;
  }
  out[(size_t)rows * 13 + r] = a;
}

// ---------------------------------------------------------------------------
// Orchestration
// ---------------------------------------------------------------------------
static inline size_t ws_align(size_t x) { return (x + 255) & ~(size_t)255; }

extern "C" void kernel_launch(void* const* d_in, const int* in_sizes, int n_in,
                              void* d_out, int out_size, void* d_ws,
                              size_t ws_size, hipStream_t stream) {
  (void)in_sizes; (void)n_in; (void)out_size; (void)ws_size;
  const int B = 8;
  const int N1 = 8192, N2 = 2048, N3 = 1024, N4 = 256, N5 = 64, N6 = 16;

  // positional inputs (setup_inputs dict insertion order)
  const float* input_pts = (const float*)d_in[1];
  const float* x6 = (const float*)d_in[2];
  const float* pts6 = (const float*)d_in[3];
  const float* x5 = (const float*)d_in[4];
  const float* pts5 = (const float*)d_in[5];
  const float* x4 = (const float*)d_in[6];
  const float* pts4 = (const float*)d_in[7];
  const float* x3 = (const float*)d_in[8];
  const float* pts3 = (const float*)d_in[9];
  const float* x2 = (const float*)d_in[10];
  const float* pts2 = (const float*)d_in[11];

  // params: JAX pytree (alphabetical) flattening of the nested params dict:
  // 12..21: bn1d_b,bn1d_g,bn2d_b,bn2d_g,bn3d_b,bn3d_g,bn4d_b,bn4d_g,bn5d_b,bn5d_g
  // 22: ccout_b  23: ccout_w
  // 24..31: cv1d {W,centers,l1b,l1w,l2b,l2w,l3b,l3w}
  // 32..39: cv2d  40..47: cv3d  48..55: cv4d  56..63: cv5d
  // 64: fcout_b  65: fcout_w
  const float* bn1d_b = (const float*)d_in[12];
  const float* bn1d_g = (const float*)d_in[13];
  const float* bn2d_b = (const float*)d_in[14];
  const float* bn2d_g = (const float*)d_in[15];
  const float* bn3d_b = (const float*)d_in[16];
  const float* bn3d_g = (const float*)d_in[17];
  const float* bn4d_b = (const float*)d_in[18];
  const float* bn4d_g = (const float*)d_in[19];
  const float* bn5d_b = (const float*)d_in[20];
  const float* bn5d_g = (const float*)d_in[21];
  const float* ccout_b = (const float*)d_in[22];
  const float* ccout_w = (const float*)d_in[23];
  struct Cv { const float *W, *cent, *l1b, *l1w, *l2b, *l2w, *l3b, *l3w; };
  auto getcv = [&](int base) {
    Cv c;
    c.W = (const float*)d_in[base + 0];
    c.cent = (const float*)d_in[base + 1];
    c.l1b = (const float*)d_in[base + 2];
    c.l1w = (const float*)d_in[base + 3];
    c.l2b = (const float*)d_in[base + 4];
    c.l2w = (const float*)d_in[base + 5];
    c.l3b = (const float*)d_in[base + 6];
    c.l3w = (const float*)d_in[base + 7];
    return c;
  };
  Cv cv1d = getcv(24), cv2d = getcv(32), cv3d = getcv(40), cv4d = getcv(48),
     cv5d = getcv(56);
  const float* fcout_b = (const float*)d_in[64];
  const float* fcout_w = (const float*)d_in[65];

  // workspace carve (~40 MB)
  char* w = (char*)d_ws;
  size_t off = 0;
  auto carve = [&](size_t bytes) {
    void* p = w + off;
    off = ws_align(off + bytes);
    return p;
  };
  int* idxbuf = (int*)carve(sizeof(int) * (size_t)B * N1 * 8);
  float* convbuf = (float*)carve(sizeof(float) * (size_t)B * N1 * 48);
  float* stats = (float*)carve(sizeof(float) * 2 * 96);
  _Float16* Whbuf = (_Float16*)carve(sizeof(_Float16) * (size_t)192 * 16 * 96);
  float* x5d = (float*)carve(sizeof(float) * (size_t)B * N5 * 192);
  float* x4d = (float*)carve(sizeof(float) * (size_t)B * N4 * 192);
  float* x3d = (float*)carve(sizeof(float) * (size_t)B * N3 * 96);
  float* x2d = (float*)carve(sizeof(float) * (size_t)B * N2 * 96);
  float* x1d = (float*)carve(sizeof(float) * (size_t)B * N1 * 48);

  // ---- L5d: x6[96] @ pts6(16) -> pts5(64), K=4, Cout=96 ----
  {
    int N = N6, M = N5, rows = B * M;
    knn_kernel<4><<<(rows + 255) / 256, 256, 0, stream>>>(pts6, pts5, idxbuf, B, N, M);
    int wn = 96 * 16 * 96;
    pack_w_kernel<<<(wn / 16 + 255) / 256, 256, 0, stream>>>(cv5d.W, Whbuf, 96 * 16, 96);
    ptconv_tile_kernel<96, 96, 4><<<rows / 16, 128, 0, stream>>>(
        x6, pts6, pts5, idxbuf, Whbuf, cv5d.cent, cv5d.l1w, cv5d.l1b, cv5d.l2w,
        cv5d.l2b, cv5d.l3w, cv5d.l3b, convbuf, B, N, M);
    bn_stats_kernel<<<96, 256, 0, stream>>>(convbuf, stats, rows, 96);
    bn_apply_concat_kernel<<<(rows * 192 + 255) / 256, 256, 0, stream>>>(
        convbuf, stats, bn5d_g, bn5d_b, x5, x5d, rows, 96, 96);
  }
  // ---- L4d: x5d[192] @ pts5(64) -> pts4(256), K=4, Cout=96 ----
  {
    int N = N5, M = N4, rows = B * M;
    knn_kernel<4><<<(rows + 255) / 256, 256, 0, stream>>>(pts5, pts4, idxbuf, B, N, M);
    int wn = 192 * 16 * 96;
    pack_w_kernel<<<(wn / 16 + 255) / 256, 256, 0, stream>>>(cv4d.W, Whbuf, 192 * 16, 96);
    ptconv_tile_kernel<192, 96, 4><<<rows / 16, 128, 0, stream>>>(
        x5d, pts5, pts4, idxbuf, Whbuf, cv4d.cent, cv4d.l1w, cv4d.l1b, cv4d.l2w,
        cv4d.l2b, cv4d.l3w, cv4d.l3b, convbuf, B, N, M);
    bn_stats_kernel<<<96, 256, 0, stream>>>(convbuf, stats, rows, 96);
    bn_apply_concat_kernel<<<(rows * 192 + 255) / 256, 256, 0, stream>>>(
        convbuf, stats, bn4d_g, bn4d_b, x4, x4d, rows, 96, 96);
  }
  // ---- L3d: x4d[192] @ pts4(256) -> pts3(1024), K=4, Cout=48 ----
  {
    int N = N4, M = N3, rows = B * M;
    knn_kernel<4><<<(rows + 255) / 256, 256, 0, stream>>>(pts4, pts3, idxbuf, B, N, M);
    int wn = 192 * 16 * 48;
    pack_w_kernel<<<(wn / 16 + 255) / 256, 256, 0, stream>>>(cv3d.W, Whbuf, 192 * 16, 48);
    ptconv_tile_kernel<192, 48, 4><<<rows / 16, 128, 0, stream>>>(
        x4d, pts4, pts3, idxbuf, Whbuf, cv3d.cent, cv3d.l1w, cv3d.l1b, cv3d.l2w,
        cv3d.l2b, cv3d.l3w, cv3d.l3b, convbuf, B, N, M);
    bn_stats_kernel<<<48, 256, 0, stream>>>(convbuf, stats, rows, 48);
    bn_apply_concat_kernel<<<(rows * 96 + 255) / 256, 256, 0, stream>>>(
        convbuf, stats, bn3d_g, bn3d_b, x3, x3d, rows, 48, 48);
  }
  // ---- L2d: x3d[96] @ pts3(1024) -> pts2(2048), K=8, Cout=48 ----
  {
    int N = N3, M = N2, rows = B * M;
    knn_kernel<8><<<(rows + 255) / 256, 256, 0, stream>>>(pts3, pts2, idxbuf, B, N, M);
    int wn = 96 * 16 * 48;
    pack_w_kernel<<<(wn / 16 + 255) / 256, 256, 0, stream>>>(cv2d.W, Whbuf, 96 * 16, 48);
    ptconv_tile_kernel<96, 48, 8><<<rows / 16, 128, 0, stream>>>(
        x3d, pts3, pts2, idxbuf, Whbuf, cv2d.cent, cv2d.l1w, cv2d.l1b, cv2d.l2w,
        cv2d.l2b, cv2d.l3w, cv2d.l3b, convbuf, B, N, M);
    bn_stats_kernel<<<48, 256, 0, stream>>>(convbuf, stats, rows, 48);
    bn_apply_concat_kernel<<<(rows * 96 + 255) / 256, 256, 0, stream>>>(
        convbuf, stats, bn2d_g, bn2d_b, x2, x2d, rows, 48, 48);
  }
  // ---- L1d: x2d[96] @ pts2(2048) -> input_pts(8192), K=8, Cout=48 ----
  {
    int N = N2, M = N1, rows = B * M;
    knn_kernel<8><<<(rows + 255) / 256, 256, 0, stream>>>(pts2, input_pts, idxbuf, B, N, M);
    int wn = 96 * 16 * 48;
    pack_w_kernel<<<(wn / 16 + 255) / 256, 256, 0, stream>>>(cv1d.W, Whbuf, 96 * 16, 48);
    ptconv_tile_kernel<96, 48, 8><<<rows / 16, 128, 0, stream>>>(
        x2d, pts2, input_pts, idxbuf, Whbuf, cv1d.cent, cv1d.l1w, cv1d.l1b,
        cv1d.l2w, cv1d.l2b, cv1d.l3w, cv1d.l3b, convbuf, B, N, M);
    bn_stats_kernel<<<48, 256, 0, stream>>>(convbuf, stats, rows, 48);
    bn_apply_concat_kernel<<<(rows * 48 + 255) / 256, 256, 0, stream>>>(
        convbuf, stats, bn1d_g, bn1d_b, nullptr, x1d, rows, 48, 0);
  }
  // ---- heads ----
  {
    int rows = B * N1;
    head_kernel<<<(rows + 255) / 256, 256, 0, stream>>>(
        x1d, fcout_w, fcout_b, ccout_w, ccout_b, (float*)d_out, rows);
  }
}
